// VLSLSTM_65884798321055
// MI455X (gfx1250) — matrix operand order repaired and checked by reading, not compiled, not run on hardware
//
#include <hip/hip_runtime.h>
#include <hip/hip_bf16.h>

// Problem constants (from reference)
#define B_    128
#define TIN_  300
#define TAR_  200
#define H_    512
#define L_    2
#define G4H   (4 * H_)

// Persistent-kernel geometry: 32 WGs x 256 threads (8 waves, wave32).
// Each WG owns 16 hidden units of BOTH layers: weights for its slice
// (2 layers x {Wih,Whh} x 4 gates x 16 units x 512 K) live in 256KB LDS.
#define NWG     32
#define UPW     16
#define THREADS 256
#define SMEM_BYTES (L_ * 2 * 4 * UPW * H_ * 2)   // 262144 bytes

typedef __attribute__((ext_vector_type(16))) __bf16 v16bf;
typedef __attribute__((ext_vector_type(8)))  float  v8f;
typedef unsigned short u16;

__device__ __forceinline__ u16 f2bf(float f) {
  __bf16 h = (__bf16)f;
  return __builtin_bit_cast(unsigned short, h);
}
__device__ __forceinline__ float bf2f(u16 u) {
  unsigned int x = ((unsigned int)u) << 16;
  return __builtin_bit_cast(float, x);
}
__device__ __forceinline__ float sigm(float x) { return 1.0f / (1.0f + __expf(-x)); }

__device__ __forceinline__ v8f wmma_bf16(v16bf a, v16bf b, v8f c) {
  return __builtin_amdgcn_wmma_f32_16x16x32_bf16(false, a, false, b, (short)0, c, false, false);
}

// A-fragment (16x32 bf16, row-major global): lane holds row (lane&15),
// K chunk base (lane>>4)*8; VGPR0-3 = K[k0..k0+7], VGPR4-7 = K[k0+16..k0+23].
__device__ __forceinline__ v16bf load_a_frag(const u16* __restrict__ A, long rowStride,
                                             int m16, int kb, int lane) {
  int row = m16 + (lane & 15);
  int k0  = kb + ((lane >> 4) << 3);
  const u16* p = A + (long)row * rowStride + k0;
  union { uint4 u[2]; v16bf v; } f;
  f.u[0] = *(const uint4*)(p);
  f.u[1] = *(const uint4*)(p + 16);
  return f.v;
}

// B-fragment (32x16 bf16) from LDS weight slice (B = W^T): lane holds weight
// row (lane&15) = output unit, K chunk base k0 = kb + (lane>>4)*8; K[k0..k0+7], K[k0+16..k0+23].
__device__ __forceinline__ v16bf load_b_frag(const u16* __restrict__ Wsm, int kb, int lane) {
  int n = lane & 15;
  int k = kb + ((lane >> 4) << 3);
  const u16* p = Wsm + n * H_ + k;
  union { uint4 u[2]; v16bf v; } f;
  f.u[0] = *(const uint4*)(p);
  f.u[1] = *(const uint4*)(p + 16);
  return f.v;
}

// ---------------- prep kernels ----------------
__global__ void vls_zero_kernel(unsigned int* __restrict__ p, long n) {
  long i = (long)blockIdx.x * blockDim.x + threadIdx.x;
  long s = (long)gridDim.x * blockDim.x;
  for (; i < n; i += s) p[i] = 0u;
}

__global__ void vls_cvt_x_kernel(const float* __restrict__ x, u16* __restrict__ xbf, long n) {
  long i = (long)blockIdx.x * blockDim.x + threadIdx.x;
  long s = (long)gridDim.x * blockDim.x;
  for (; i < n; i += s) xbf[i] = f2bf(x[i]);
}

// ---------------- persistent LSTM kernel ----------------
__global__ void __launch_bounds__(THREADS)
vls_lstm_persistent(const u16* __restrict__ xbf,       // [B][TIN][H] bf16
                    const int* __restrict__ len_in,    // [B]
                    const int* __restrict__ len_ar,    // [B]
                    const float* __restrict__ Wih,     // [L][4H][H]
                    const float* __restrict__ Whh,     // [L][4H][H]
                    const float* __restrict__ bih,     // [L][4H]
                    const float* __restrict__ bhh,     // [L][4H]
                    u16* __restrict__ x1bf,            // [B][H]  layer-1 input (unmasked hn0)
                    u16* __restrict__ xdec,            // [B][H]  decoder feedback input
                    u16* __restrict__ hbf,             // [L][2][B][H] double-buffered h carry
                    float* __restrict__ cst,           // [L][B][H]    c carry (owner-private)
                    unsigned int* __restrict__ barcnt, // grid barrier counter
                    float* __restrict__ out)           // teafo [B][TIN][H] ++ aureg [B][TAR][H]
{
  extern __shared__ u16 smem[];   // [l][mat][gate][unit(16)][H] bf16
  const int tid  = threadIdx.x;
  const int lane = tid & 31;
  const int wave = tid >> 5;
  const int u0   = blockIdx.x * UPW;

  // ---- stage this WG's weight slice into LDS (f32 -> bf16), 1 row / thread ----
  {
    int unit = tid & 15;
    int g    = (tid >> 4) & 3;
    int mat  = (tid >> 6) & 1;
    int l    = (tid >> 7) & 1;
    const float* src = (mat == 0 ? Wih : Whh) + ((long)l * G4H + g * H_ + (u0 + unit)) * (long)H_;
    u16* dst = smem + (long)tid * H_;
    for (int k = 0; k < H_; k += 4) {
      float4 v = *(const float4*)(src + k);
      dst[k + 0] = f2bf(v.x); dst[k + 1] = f2bf(v.y);
      dst[k + 2] = f2bf(v.z); dst[k + 3] = f2bf(v.w);
    }
  }
  __syncthreads();

  // ---- per-lane step-invariant values ----
  const int m16  = wave * 16;         // batch tile base (8 waves x 16 = 128)
  const int hi   = lane >> 4;
  const int ncol = u0 + (lane & 15);  // owned hidden unit (pointwise column)
  float bias_[L_][4];
  #pragma unroll
  for (int l = 0; l < L_; ++l)
    #pragma unroll
    for (int g = 0; g < 4; ++g)
      bias_[l][g] = bih[l * G4H + g * H_ + ncol] + bhh[l * G4H + g * H_ + ncol];
  int lin[8], lar[8];
  #pragma unroll
  for (int r = 0; r < 8; ++r) {
    int b = m16 + r + 8 * hi;
    lin[r] = len_in[b];
    lar[r] = len_ar[b];
  }

  unsigned int gen = 0;

  for (int s = 0; s < TIN_ + TAR_; ++s) {
    const bool enc = (s < TIN_);
    const int  t   = enc ? s : (s - TIN_);
    const int  rp  = s & 1, wp = rp ^ 1;

    for (int l = 0; l < L_; ++l) {
      // ---- GEMM: gates[128 x 64] = Ain@Wih_sl^T + Hprev@Whh_sl^T ----
      const u16* Ain; long astr_in;
      if (l == 0) {
        if (enc) { Ain = xbf + (long)t * H_; astr_in = (long)TIN_ * H_; }
        else     { Ain = xdec;               astr_in = H_; }
      } else     { Ain = x1bf;               astr_in = H_; }
      const u16* Ahh = hbf + (long)(l * 2 + rp) * (B_ * H_);

      v8f acc[4] = {};
      const u16* wl = smem + (long)(l * 2) * 4 * UPW * H_;
      #pragma unroll
      for (int src = 0; src < 2; ++src) {
        const u16* A    = (src == 0) ? Ain : Ahh;
        long       astr = (src == 0) ? astr_in : (long)H_;
        const u16* wm   = wl + (long)src * 4 * UPW * H_;
        for (int kb = 0; kb < H_; kb += 32) {
          v16bf a = load_a_frag(A, astr, m16, kb, lane);
          #pragma unroll
          for (int g = 0; g < 4; ++g) {
            v16bf b = load_b_frag(wm + (long)g * UPW * H_, kb, lane);
            acc[g] = wmma_bf16(a, b, acc[g]);
          }
        }
      }

      // ---- pointwise gate math + state update ----
      float* cl = cst + (long)l * B_ * H_;
      u16*   hw = hbf + (long)(l * 2 + wp) * (B_ * H_);
      #pragma unroll
      for (int r = 0; r < 8; ++r) {
        int  brow  = m16 + r + 8 * hi;
        long idx   = (long)brow * H_ + ncol;
        bool valid = t < (enc ? lin[r] : lar[r]);
        float gi = sigm(acc[0][r] + bias_[l][0]);
        float gf = sigm(acc[1][r] + bias_[l][1]);
        float gg = tanhf(acc[2][r] + bias_[l][2]);
        float go = sigm(acc[3][r] + bias_[l][3]);
        float cold = cl[idx];
        float cn   = gf * cold + gi * gg;
        float hn   = go * tanhf(cn);
        float hold = bf2f(Ahh[idx]);
        cl[idx] = valid ? cn : cold;
        hw[idx] = f2bf(valid ? hn : hold);
        if (l == 0) {
          x1bf[idx] = f2bf(hn);                 // unmasked hn feeds layer 1
        } else {
          float ov = valid ? hn : 0.0f;         // masked top output
          if (enc) {
            out[((long)brow * TIN_ + t) * H_ + ncol] = ov;
            if (t == lin[r] - 1) xdec[idx] = f2bf(ov);   // decoder init input
          } else {
            out[(long)B_ * TIN_ * H_ + ((long)brow * TAR_ + t) * H_ + ncol] = ov;
            xdec[idx] = f2bf(ov);               // masked output feeds next step
          }
        }
      }

      // ---- grid-wide barrier (generation counter) ----
      __syncthreads();
      ++gen;
      if (tid == 0) {
        __threadfence();
        atomicAdd(barcnt, 1u);
        while (__hip_atomic_load(barcnt, __ATOMIC_ACQUIRE, __HIP_MEMORY_SCOPE_AGENT)
               < gen * (unsigned)NWG) {
          __builtin_amdgcn_s_sleep(1);
        }
      }
      __syncthreads();
    }
  }
}

// ---------------- host entry ----------------
extern "C" void kernel_launch(void* const* d_in, const int* in_sizes, int n_in,
                              void* d_out, int out_size, void* d_ws, size_t ws_size,
                              hipStream_t stream) {
  (void)in_sizes; (void)n_in; (void)out_size; (void)ws_size;
  const float* x      = (const float*)d_in[0];
  const int*   len_in = (const int*)d_in[1];
  const int*   len_ar = (const int*)d_in[2];
  /* d_in[3] = mask_aureg (bool) -- recomputed from len_ar on device */
  const float* Wih = (const float*)d_in[4];
  const float* Whh = (const float*)d_in[5];
  const float* bih = (const float*)d_in[6];
  const float* bhh = (const float*)d_in[7];
  float* out = (float*)d_out;

  char* ws = (char*)d_ws;
  size_t o = 0;
  u16* xbf  = (u16*)(ws + o); o += (size_t)B_ * TIN_ * H_ * 2;   // 39.3 MB
  u16* x1bf = (u16*)(ws + o); o += (size_t)B_ * H_ * 2;
  u16* xdec = (u16*)(ws + o); o += (size_t)B_ * H_ * 2;
  // zero-initialized region: h (double-buffered), c, barrier counter
  char* zbase = ws + o;
  u16*   hbf    = (u16*)(ws + o);  o += (size_t)L_ * 2 * B_ * H_ * 2;
  float* cst    = (float*)(ws + o); o += (size_t)L_ * B_ * H_ * 4;
  unsigned int* barcnt = (unsigned int*)(ws + o); o += 64;
  long zwords = (long)((ws + o) - zbase) / 4;

  vls_zero_kernel<<<256, THREADS, 0, stream>>>((unsigned int*)zbase, zwords);
  vls_cvt_x_kernel<<<2048, THREADS, 0, stream>>>(x, xbf, (long)B_ * TIN_ * H_);
  vls_lstm_persistent<<<NWG, THREADS, SMEM_BYTES, stream>>>(
      xbf, len_in, len_ar, Wih, Whh, bih, bhh,
      x1bf, xdec, hbf, cst, barcnt, out);
}
